// GraphSAGERegressor_16372415332826
// MI455X (gfx1250) — matrix operand order repaired and checked by reading, not compiled, not run on hardware
//
#include <hip/hip_runtime.h>
#include <hip/hip_bf16.h>

// ---------------------------------------------------------------------------
// GraphSAGE regressor for MI455X (gfx1250, wave32, WMMA)
//   h1 = ReLU([mean1|x]   @ [W1l|W1r]^T + b1)      K=256 f16 WMMA GEMM
//   h2 = ReLU([mean2|h1]  @ [W2l|W2r]^T + b2)      K=512 f16 WMMA GEMM
//   out = h2 @ head_w^T + head_b
// Edge aggregation via coalesced f32 global atomics (memory-bound phase).
// GEMM stages B fragments in LDS (double-buffered, 1 barrier per K-chunk).
// ---------------------------------------------------------------------------

#define NN 50000        // N_NODES
#define DH 256          // hidden dim

typedef __attribute__((ext_vector_type(16))) _Float16     v16h;
typedef __attribute__((ext_vector_type(8)))  float        v8f;
typedef __attribute__((ext_vector_type(4)))  unsigned int u32x4;

union FragU { u32x4 u[2]; v16h v; };   // 32 bytes; trivially constructible

static __device__ __forceinline__ unsigned short f2h_bits(float f) {
    _Float16 h = (_Float16)f;
    return __builtin_bit_cast(unsigned short, h);
}

// ---------------------------------------------------------------- zero fill
__global__ void zero_f32(float* __restrict__ p, size_t n) {
    size_t i = (size_t)blockIdx.x * blockDim.x + threadIdx.x;
    size_t s = (size_t)gridDim.x * blockDim.x;
    for (; i < n; i += s) p[i] = 0.0f;
}

// ---------------------------------------------------------------- degree
__global__ void degree_kernel(const int* __restrict__ ei, int E,
                              float* __restrict__ deg) {
    int e = blockIdx.x * blockDim.x + threadIdx.x;
    if (e >= E) return;
    int dst = ei[E + e];
    __hip_atomic_fetch_add(&deg[dst], 1.0f, __ATOMIC_RELAXED,
                           __HIP_MEMORY_SCOPE_AGENT);
}

// ------------------------------------------------- edge gather/scatter-add
// thread = (edge, float4-chunk), warp covers one edge's contiguous row
// segment -> coalesced 512B gathers + contiguous atomic bursts.
// lgc4 = log2(d/4); d = 4 << lgc4.
__global__ void scatter_kernel(const float* __restrict__ feat,
                               const int* __restrict__ ei, int E, int lgc4,
                               float* __restrict__ agg) {
    unsigned tid = blockIdx.x * blockDim.x + threadIdx.x;
    unsigned total = (unsigned)E << lgc4;
    if (tid >= total) return;
    int e = (int)(tid >> lgc4);
    int c = (int)(tid & ((1u << lgc4) - 1u));
    int d = 4 << lgc4;
    int src = ei[e];
    int dst = ei[E + e];
    const float4* xs = (const float4*)(feat + (size_t)src * d);
    float4 v = xs[c];
    float* ab = agg + (size_t)dst * d + (size_t)c * 4;
    __hip_atomic_fetch_add(ab + 0, v.x, __ATOMIC_RELAXED, __HIP_MEMORY_SCOPE_AGENT);
    __hip_atomic_fetch_add(ab + 1, v.y, __ATOMIC_RELAXED, __HIP_MEMORY_SCOPE_AGENT);
    __hip_atomic_fetch_add(ab + 2, v.z, __ATOMIC_RELAXED, __HIP_MEMORY_SCOPE_AGENT);
    __hip_atomic_fetch_add(ab + 3, v.w, __ATOMIC_RELAXED, __HIP_MEMORY_SCOPE_AGENT);
}

// ------------------------------------- build Xin = [agg/deg | self] in f16
__global__ void prep_kernel(const float* __restrict__ agg,
                            const float* __restrict__ deg,
                            const float* __restrict__ self,
                            int dmean, int K,
                            unsigned short* __restrict__ Xin) {
    size_t tid = (size_t)blockIdx.x * blockDim.x + threadIdx.x;
    size_t total = (size_t)NN * K;
    if (tid >= total) return;
    int row = (int)(tid / K);
    int col = (int)(tid % K);
    float v;
    if (col < dmean) {
        float d = deg[row];
        d = d < 1.0f ? 1.0f : d;
        v = agg[(size_t)row * dmean + col] / d;
    } else {
        int dself = K - dmean;
        v = self[(size_t)row * dself + (col - dmean)];
    }
    Xin[tid] = f2h_bits(v);
}

// ---------------------------------------------- pack W = [Wl|Wr] (DH x K)
// into per-lane B fragments for v_wmma_f32_16x16x32_f16.
// B layout (32x16, K x N): lanes 0-15 -> N=lane, K = kc*32 + 0..15 (halves
// 0..15 contiguous); lanes 16-31 -> N=lane-16, K = kc*32 + 16..31.
// packB[((kc*16 + nt)*32 + lane)*16 + h]
__global__ void pack_weights(const float* __restrict__ Wl,
                             const float* __restrict__ Wr,
                             int Kin, int KC,
                             unsigned short* __restrict__ packB) {
    int tid = blockIdx.x * blockDim.x + threadIdx.x;
    int total = KC * 16 * 32;
    if (tid >= total) return;
    int lane = tid & 31;
    int nt   = (tid >> 5) & 15;
    int kc   = tid >> 9;
    int col  = nt * 16 + (lane & 15);
    int kbase = kc * 32 + ((lane >> 4) * 16);
    unsigned short* dst = packB + (size_t)tid * 16;
#pragma unroll
    for (int h = 0; h < 16; ++h) {
        int kg = kbase + h;
        float w = (kg < Kin) ? Wl[(size_t)col * Kin + kg]
                             : Wr[(size_t)col * Kin + (kg - Kin)];
        dst[h] = f2h_bits(w);
    }
}

// ------------------------------------------------------- WMMA GEMM + ReLU
// out[NN][DH] = ReLU(Xin[NN][K] @ Wc^T + bias). 8 waves/block; each wave
// owns a 16-row tile and all 256 output cols (16 N-tiles, v8f accum each).
// B fragments for the current K-chunk staged in LDS, double-buffered:
// next chunk's 16KB global->LDS copy overlaps the current chunk's 16 WMMAs,
// one s_barrier per chunk.
template <int K>
__global__ __launch_bounds__(256)
void sage_gemm_wmma(const unsigned short* __restrict__ Xin,
                    const unsigned short* __restrict__ packB,
                    const float* __restrict__ bias,
                    float* __restrict__ out) {
    constexpr int KC = K / 32;
    constexpr int CH = 16 * 32 * 16;            // halves per K-chunk (16KB)
    __shared__ unsigned short sB[2][CH];        // 32KB double buffer

    const int lane = threadIdx.x & 31;
    const int wave = threadIdx.x >> 5;
    int tile = blockIdx.x * 8 + wave;           // 16-row tile index
    const int maxTile = NN / 16 - 1;
    if (tile > maxTile) tile = maxTile;         // keep all waves in barriers
                                                // (dup waves write identical data)

    // A fragment addressing (contiguous k order used here, same as packB):
    //   lanes 0-15: M=lane,    halves 0..15 -> K 0..15
    //   lanes16-31: M=lane-16, halves 0..15 -> K 16..31
    const int rowA = lane & 15;
    const int kbA  = (lane >> 4) * 16;
    const unsigned short* arow = Xin + (size_t)(tile * 16 + rowA) * K + kbA;

    v8f acc[16] = {};

    // prologue: stage chunk 0
    {
        const u32x4* g = (const u32x4*)packB;
        u32x4* l = (u32x4*)sB[0];
#pragma unroll
        for (int i = 0; i < 4; ++i)
            l[threadIdx.x + 256 * i] = g[threadIdx.x + 256 * i];
    }
    __syncthreads();

    for (int kc = 0; kc < KC; ++kc) {
        const int cur = kc & 1;

        // stage next chunk into the other buffer (overlaps WMMA below)
        if (kc + 1 < KC) {
            const u32x4* g = (const u32x4*)(packB + (size_t)(kc + 1) * CH);
            u32x4* l = (u32x4*)sB[cur ^ 1];
#pragma unroll
            for (int i = 0; i < 4; ++i)
                l[threadIdx.x + 256 * i] = g[threadIdx.x + 256 * i];
            __builtin_prefetch(arow + (kc + 1) * 32, 0, 0);  // next A chunk
        }

        FragU A;
        const u32x4* ap = (const u32x4*)(arow + kc * 32);
        A.u[0] = ap[0];   // halves 0..7  : K = kc*32 + kbA .. +7
        A.u[1] = ap[1];   // halves 8..15 : K = kc*32 + kbA + 8 .. +15

        const unsigned short* bbase = &sB[cur][lane * 16];
#pragma unroll
        for (int nt = 0; nt < 16; ++nt) {
            FragU B;
            const u32x4* bp = (const u32x4*)(bbase + nt * 512);
            B.u[0] = bp[0];
            B.u[1] = bp[1];
            acc[nt] = __builtin_amdgcn_wmma_f32_16x16x32_f16(
                false, A.v, false, B.v, (short)0, acc[nt], false, false);
        }
        __syncthreads();   // everyone done with sB[cur]; sB[cur^1] filled
    }

    // C/D layout: VGPR r -> M = r + 8*(lane>=16), N = lane & 15
    const int mbase = (lane >> 4) * 8;
    const int colL  = lane & 15;
#pragma unroll
    for (int nt = 0; nt < 16; ++nt) {
        const int col = nt * 16 + colL;
        const float bv = bias[col];
#pragma unroll
        for (int r = 0; r < 8; ++r) {
            float v = acc[nt][r] + bv;
            v = v > 0.0f ? v : 0.0f;
            out[(size_t)(tile * 16 + mbase + r) * DH + col] = v;
        }
    }
}

// -------------------------------------------------------------- head (dot)
__global__ __launch_bounds__(256)
void head_kernel(const float* __restrict__ h2, const float* __restrict__ hw,
                 const float* __restrict__ hb, float* __restrict__ out) {
    int node = (int)(((size_t)blockIdx.x * blockDim.x + threadIdx.x) >> 5);
    int lane = threadIdx.x & 31;
    if (node >= NN) return;
    const float4* r = (const float4*)(h2 + (size_t)node * DH);
    const float4* w = (const float4*)hw;
    float s = 0.0f;
#pragma unroll 2
    for (int i = lane; i < DH / 4; i += 32) {
        float4 a = r[i];
        float4 b = w[i];
        s += a.x * b.x + a.y * b.y + a.z * b.z + a.w * b.w;
    }
#pragma unroll
    for (int off = 16; off > 0; off >>= 1) s += __shfl_xor(s, off, 32);
    if (lane == 0) out[node] = s + hb[0];
}

// ---------------------------------------------------------------------------
extern "C" void kernel_launch(void* const* d_in, const int* in_sizes, int n_in,
                              void* d_out, int out_size, void* d_ws,
                              size_t ws_size, hipStream_t stream) {
    (void)n_in; (void)out_size; (void)ws_size;

    const float* x   = (const float*)d_in[0];
    const int*   ei  = (const int*)d_in[1];
    const float* W1l = (const float*)d_in[2];
    const float* b1  = (const float*)d_in[3];
    const float* W1r = (const float*)d_in[4];
    const float* W2l = (const float*)d_in[5];
    const float* b2  = (const float*)d_in[6];
    const float* W2r = (const float*)d_in[7];
    const float* hw  = (const float*)d_in[8];
    const float* hb  = (const float*)d_in[9];
    float* out = (float*)d_out;

    const int E = in_sizes[1] / 2;

    // -------- workspace carve-out (256B aligned)
    char* ws = (char*)d_ws;
    size_t off = 0;
    auto carve = [&](size_t bytes) -> char* {
        char* p = ws + off;
        off = (off + bytes + 255) & ~(size_t)255;
        return p;
    };
    float* deg  = (float*)carve((size_t)NN * 4);
    float* agg1 = (float*)carve((size_t)NN * 128 * 4);
    float* agg2 = (float*)carve((size_t)NN * 256 * 4);
    float* h1   = (float*)carve((size_t)NN * 256 * 4);
    float* h2   = (float*)carve((size_t)NN * 256 * 4);
    unsigned short* Xin1 = (unsigned short*)carve((size_t)NN * 256 * 2);
    unsigned short* Xin2 = (unsigned short*)carve((size_t)NN * 512 * 2);
    unsigned short* pB1  = (unsigned short*)carve((size_t)8  * 16 * 32 * 16 * 2);
    unsigned short* pB2  = (unsigned short*)carve((size_t)16 * 16 * 32 * 16 * 2);

    // -------- init (must re-zero every call: graph replay reuses ws)
    zero_f32<<<1024, 256, 0, stream>>>(deg,  (size_t)NN);
    zero_f32<<<2048, 256, 0, stream>>>(agg1, (size_t)NN * 128);
    zero_f32<<<2048, 256, 0, stream>>>(agg2, (size_t)NN * 256);

    degree_kernel<<<(E + 255) / 256, 256, 0, stream>>>(ei, E, deg);
    pack_weights<<<(8 * 16 * 32 + 255) / 256, 256, 0, stream>>>(W1l, W1r, 128, 8,  pB1);
    pack_weights<<<(16 * 16 * 32 + 255) / 256, 256, 0, stream>>>(W2l, W2r, 256, 16, pB2);

    // -------- layer 1  (d=128 -> lgc4=5)
    {
        unsigned t = (unsigned)E << 5;
        scatter_kernel<<<(t + 255) / 256, 256, 0, stream>>>(x, ei, E, 5, agg1);
        size_t p = (size_t)NN * 256;
        prep_kernel<<<(unsigned)((p + 255) / 256), 256, 0, stream>>>(
            agg1, deg, x, 128, 256, Xin1);
        sage_gemm_wmma<256><<<(NN / 16 + 7) / 8, 256, 0, stream>>>(
            Xin1, pB1, b1, h1);
    }

    // -------- layer 2  (d=256 -> lgc4=6)
    {
        unsigned t = (unsigned)E << 6;
        scatter_kernel<<<(t + 255) / 256, 256, 0, stream>>>(h1, ei, E, 6, agg2);
        size_t p = (size_t)NN * 512;
        prep_kernel<<<(unsigned)((p + 255) / 256), 256, 0, stream>>>(
            agg2, deg, h1, 256, 512, Xin2);
        sage_gemm_wmma<512><<<(NN / 16 + 7) / 8, 256, 0, stream>>>(
            Xin2, pB2, b2, h2);
    }

    // -------- head
    head_kernel<<<(NN * 32 + 255) / 256, 256, 0, stream>>>(h2, hw, hb, out);
}
